// GNN_3126736192020
// MI455X (gfx1250) — matrix operand order repaired and checked by reading, not compiled, not run on hardware
//
#include <hip/hip_runtime.h>

typedef __attribute__((ext_vector_type(16))) _Float16 v16h;
typedef __attribute__((ext_vector_type(8)))  float    v8f;

#define HD 32
#define KCHUNKS 33              // 32 K-chunks from w2 + 1 bias chunk (b2)
#define WPACK_PER_LAYER (KCHUNKS * 2 * 32 * 16)   // 33792 f16 elements (66 KB)
#define HID_STRIDE 20           // dwords per edge in packed hid buffer (80 B)
#define TILES 4                 // 16-edge WMMA tiles per wave (64 edges)

// ---------------------------------------------------------------------------
// Node encoder: h = x @ enc_w + enc_b   [N,8]->[N,32], f32 + packed f16 copy
// ---------------------------------------------------------------------------
__global__ void __launch_bounds__(256) k_encoder(const float* __restrict__ x,
    const float* __restrict__ encw, const float* __restrict__ encb,
    float* __restrict__ h, _Float16* __restrict__ hh, int N)
{
  int n = blockIdx.x * blockDim.x + threadIdx.x;
  if (n >= N) return;
  float xi[8];
#pragma unroll
  for (int c = 0; c < 8; ++c) xi[c] = x[(size_t)n * 8 + c];
  float sv[HD];
#pragma unroll 4
  for (int j = 0; j < HD; ++j) {
    float s = encb[j];
#pragma unroll
    for (int c = 0; c < 8; ++c) s += xi[c] * encw[c * HD + j];
    sv[j] = s;
    h[(size_t)n * HD + j] = s;
  }
  uint* hp = (uint*)(hh + (size_t)n * HD);
#pragma unroll
  for (int d = 0; d < HD / 2; ++d) {
    union { _Float16 f[2]; uint u; } pk;
    pk.f[0] = (_Float16)sv[2 * d];
    pk.f[1] = (_Float16)sv[2 * d + 1];
    hp[d] = pk.u;
  }
}

// ---------------------------------------------------------------------------
// In-degree (for scatter-mean normalization)
// ---------------------------------------------------------------------------
__global__ void __launch_bounds__(256) k_deg(const int* __restrict__ edge_index,
    float* __restrict__ deg, int E)
{
  int e = blockIdx.x * blockDim.x + threadIdx.x;
  if (e < E) atomicAdd(&deg[edge_index[E + e]], 1.0f);
}

// ---------------------------------------------------------------------------
// Per-layer edge MLP stage 1: hid = relu(ea @ w1 + b1), stored as packed f16
// pairs: dword d = (hid[2d], hid[2d+1]) for d<16, dword 16 = (1.0, 0) bias.
// ---------------------------------------------------------------------------
__global__ void __launch_bounds__(256) k_edgemlp(const float* __restrict__ edge_attr,
    const float* __restrict__ w1l, const float* __restrict__ b1l,
    uint* __restrict__ hidp, int E)
{
  int e = blockIdx.x * blockDim.x + threadIdx.x;
  if (e >= E) return;
  const float4 ea = *(const float4*)(edge_attr + (size_t)e * 4);
  float hv[HD];
#pragma unroll 4
  for (int k = 0; k < HD; ++k) {
    float s = b1l[k]
            + ea.x * w1l[0 * HD + k] + ea.y * w1l[1 * HD + k]
            + ea.z * w1l[2 * HD + k] + ea.w * w1l[3 * HD + k];
    hv[k] = fmaxf(s, 0.0f);
  }
  uint* o = hidp + (size_t)e * HID_STRIDE;
#pragma unroll
  for (int d = 0; d < HD / 2; ++d) {
    union { _Float16 f[2]; uint u; } pk;
    pk.f[0] = (_Float16)hv[2 * d];
    pk.f[1] = (_Float16)hv[2 * d + 1];
    o[d] = pk.u;
  }
  union { _Float16 f[2]; uint u; } pb;
  pb.f[0] = (_Float16)1.0f; pb.f[1] = (_Float16)0.0f;
  o[16] = pb.u;
}

// ---------------------------------------------------------------------------
// Pack W = [w2 reshaped ; b2] into the 16-bit B-matrix (32x16/chunk) layout.
//   B elem s, lanes 0-15 : K = s (s<8), s+8 (s>=8) | lanes 16-31 : K = s+8 (s<8), s+16 (s>=8)
//   kc <  32 : W[32*kc+Koff][N] = w2[kc][Koff*32 + N]
//   kc == 32 : W[.][N]          = b2[Koff*32 + N]     (bias chunk)
// ---------------------------------------------------------------------------
__global__ void __launch_bounds__(256) k_pack(const float* __restrict__ w2,
    const float* __restrict__ b2, _Float16* __restrict__ wp, int total)
{
  int t = blockIdx.x * blockDim.x + threadIdx.x;
  if (t >= total) return;
  int slot = t & 15;
  int r = t >> 4;
  int lane = r & 31; r >>= 5;
  int nh   = r & 1;  r >>= 1;
  int kc   = r % KCHUNKS;
  int lay  = r / KCHUNKS;
  int Koff = (slot & 7) + ((lane >= 16) ? 8 : 0) + ((slot >> 3) << 4);
  int Nn   = nh * 16 + (lane & 15);
  float v = (kc < HD)
      ? w2[(size_t)lay * (HD * HD * HD) + (size_t)kc * (HD * HD) + Koff * HD + Nn]
      : b2[(size_t)lay * (HD * HD) + Koff * HD + Nn];
  wp[t] = (_Float16)v;
}

// ---------------------------------------------------------------------------
// Fused edge GEMM. Per wave, 64 edges = four 16-edge WMMA tiles:
//   msg = [outer(hid,h_src) ; h_src] @ Wpack     (264 x v_wmma f16->f32)
//   agg[dst] += msg                              (f32 L2 atomics)
// Per K-chunk: 4x ds_load_b128 (B panel, LDS) + 32x v_pk_mul_f16 (rank-1 A
// tiles) + 8x WMMA — B loads amortized over 4 tiles, and >6 independent
// WMMAs separate each av write from its reader (kills the 4-NOP hazard).
// ---------------------------------------------------------------------------
__global__ void __launch_bounds__(256) k_edge_gemm(
    const _Float16* __restrict__ hsrch,   // [N][32] f16 node features
    const int*      __restrict__ edge_index,
    const uint*     __restrict__ hidp,    // [E][20] packed hid dwords
    const _Float16* __restrict__ wpackl,  // [33][2][32][16]
    float* __restrict__ agg,
    int E)
{
  __shared__ __attribute__((aligned(32))) _Float16 ldsw[WPACK_PER_LAYER];
  {
    const uint4* g = (const uint4*)wpackl;
    uint4*       l = (uint4*)ldsw;
    const int nvec = WPACK_PER_LAYER / 8;
    for (int i = threadIdx.x; i < nvec; i += blockDim.x) l[i] = g[i];
  }
  __syncthreads();

  const int lane = threadIdx.x & 31;
  const int half = lane >> 4;
  const int mrow = lane & 15;
  const int wid  = (blockIdx.x * blockDim.x + threadIdx.x) >> 5;
  const int nw   = (gridDim.x * blockDim.x) >> 5;
  const int ngrp = (E + (16 * TILES - 1)) / (16 * TILES);

  const v16h* lwp = (const v16h*)ldsw;

  for (int t = wid; t < ngrp; t += nw) {
    int dsts[TILES];
    union HidU { uint u[17]; _Float16 h[34]; } hid[TILES];
    v16h hsv[TILES];

#pragma unroll
    for (int tt = 0; tt < TILES; ++tt) {
      const int  e     = t * (16 * TILES) + tt * 16 + mrow;
      const bool valid = (e < E);
      const int  ec    = valid ? e : 0;
      const int  src   = edge_index[ec];
      dsts[tt]         = edge_index[E + ec];

      // packed hid: 4 x b128 + 1 x b32
      const uint*  hpE = hidp + (size_t)ec * HID_STRIDE;
      const uint4* h4  = (const uint4*)hpE;
      const uint4  qa = h4[0], qb = h4[1], qc = h4[2], qd = h4[3];
      hid[tt].u[0]=qa.x;  hid[tt].u[1]=qa.y;  hid[tt].u[2]=qa.z;  hid[tt].u[3]=qa.w;
      hid[tt].u[4]=qb.x;  hid[tt].u[5]=qb.y;  hid[tt].u[6]=qb.z;  hid[tt].u[7]=qb.w;
      hid[tt].u[8]=qc.x;  hid[tt].u[9]=qc.y;  hid[tt].u[10]=qc.z; hid[tt].u[11]=qc.w;
      hid[tt].u[12]=qd.x; hid[tt].u[13]=qd.y; hid[tt].u[14]=qd.z; hid[tt].u[15]=qd.w;
      hid[tt].u[16]=hpE[16];

      // h_src f16, A-layout K-runs: i = {0..7,16..23} + 8*half -> 2 x b128
      const _Float16* hb = hsrch + (size_t)src * HD;
      union { uint4 q[2]; v16h v; } hs;
      hs.q[0] = *(const uint4*)(hb + 8 * half);
      hs.q[1] = *(const uint4*)(hb + 16 + 8 * half);
      const _Float16 vm = (_Float16)(valid ? 1.0f : 0.0f);
      hsv[tt] = hs.v * vm;
    }

    // compiler barrier: keep per-chunk LDS B loads inside this loop
    asm volatile("" ::: "memory");

    v8f accA[TILES];   // msg cols 0..15
    v8f accB[TILES];   // msg cols 16..31
#pragma unroll
    for (int tt = 0; tt < TILES; ++tt) { accA[tt] = (v8f){}; accB[tt] = (v8f){}; }

#pragma unroll
    for (int kc = 0; kc < KCHUNKS; ++kc) {
      const v16h b0 = lwp[(kc * 2 + 0) * 32 + lane];
      const v16h b1 = lwp[(kc * 2 + 1) * 32 + lane];
#pragma unroll
      for (int tt = 0; tt < TILES; ++tt) {
        const v16h av = hsv[tt] * hid[tt].h[kc];
        accA[tt] = __builtin_amdgcn_wmma_f32_16x16x32_f16(false, av, false, b0,
                                                          (short)0, accA[tt], false, false);
        accB[tt] = __builtin_amdgcn_wmma_f32_16x16x32_f16(false, av, false, b1,
                                                          (short)0, accB[tt], false, false);
      }
    }

    // scatter-add: C layout lane -> (M = r + 8*half, N = lane & 15)
    const int j = mrow;
#pragma unroll
    for (int tt = 0; tt < TILES; ++tt) {
#pragma unroll
      for (int r = 0; r < 8; ++r) {
        const int m  = r + 8 * half;
        const int dm = __shfl(dsts[tt], m, 32);
        atomicAdd(&agg[(size_t)dm * HD + j],      accA[tt][r]);
        atomicAdd(&agg[(size_t)dm * HD + 16 + j], accB[tt][r]);
      }
    }
  }
}

// ---------------------------------------------------------------------------
// NNConv combine: h' = relu(agg/deg + h @ root_w + conv_b), f32 + f16 copy
// ---------------------------------------------------------------------------
__global__ void __launch_bounds__(256) k_update(const float* __restrict__ hin,
    const float* __restrict__ agg, const float* __restrict__ deg,
    const float* __restrict__ rootw, const float* __restrict__ convb,
    float* __restrict__ hout, _Float16* __restrict__ houth, int N)
{
  int n = blockIdx.x * blockDim.x + threadIdx.x;
  if (n >= N) return;
  const float inv = 1.0f / fmaxf(deg[n], 1.0f);
  float hi[HD];
#pragma unroll
  for (int i = 0; i < HD; ++i) hi[i] = hin[(size_t)n * HD + i];
  float sv[HD];
#pragma unroll 4
  for (int j = 0; j < HD; ++j) {
    float s = agg[(size_t)n * HD + j] * inv + convb[j];
#pragma unroll
    for (int i = 0; i < HD; ++i) s += hi[i] * rootw[i * HD + j];
    s = fmaxf(s, 0.0f);
    sv[j] = s;
    hout[(size_t)n * HD + j] = s;
  }
  uint* hp = (uint*)(houth + (size_t)n * HD);
#pragma unroll
  for (int d = 0; d < HD / 2; ++d) {
    union { _Float16 f[2]; uint u; } pk;
    pk.f[0] = (_Float16)sv[2 * d];
    pk.f[1] = (_Float16)sv[2 * d + 1];
    hp[d] = pk.u;
  }
}

// ---------------------------------------------------------------------------
// Output head: out = h @ out_w + out_b     [N,32] -> [N,3]
// ---------------------------------------------------------------------------
__global__ void __launch_bounds__(256) k_out(const float* __restrict__ h,
    const float* __restrict__ ow, const float* __restrict__ ob,
    float* __restrict__ out, int N, int OD)
{
  int n = blockIdx.x * blockDim.x + threadIdx.x;
  if (n >= N) return;
  float hi[HD];
#pragma unroll
  for (int i = 0; i < HD; ++i) hi[i] = h[(size_t)n * HD + i];
  for (int o = 0; o < OD; ++o) {
    float s = ob[o];
#pragma unroll
    for (int i = 0; i < HD; ++i) s += hi[i] * ow[i * OD + o];
    out[(size_t)n * OD + o] = s;
  }
}

// ---------------------------------------------------------------------------
extern "C" void kernel_launch(void* const* d_in, const int* in_sizes, int n_in,
                              void* d_out, int out_size, void* d_ws, size_t ws_size,
                              hipStream_t stream)
{
  (void)n_in; (void)out_size; (void)ws_size;
  const float* x          = (const float*)d_in[0];
  const int*   edge_index = (const int*)  d_in[1];
  const float* edge_attr  = (const float*)d_in[2];
  // d_in[3] = batch (unused: single graph, per-node output)
  const float* enc_w      = (const float*)d_in[4];
  const float* enc_b      = (const float*)d_in[5];
  const float* w1         = (const float*)d_in[6];
  const float* b1         = (const float*)d_in[7];
  const float* w2         = (const float*)d_in[8];
  const float* b2         = (const float*)d_in[9];
  const float* root_w     = (const float*)d_in[10];
  const float* conv_b     = (const float*)d_in[11];
  const float* out_w      = (const float*)d_in[12];
  const float* out_b      = (const float*)d_in[13];

  const int N  = in_sizes[0] / 8;
  const int E  = in_sizes[2] / 4;
  const int L  = in_sizes[6] / (4 * HD);
  const int OD = in_sizes[13];

  // workspace carve-up (~84 MB)
  char* p = (char*)d_ws;
  float*    h0   = (float*)p;    p += (size_t)N * HD * sizeof(float);
  float*    h1   = (float*)p;    p += (size_t)N * HD * sizeof(float);
  float*    agg  = (float*)p;    p += (size_t)N * HD * sizeof(float);
  _Float16* h0h  = (_Float16*)p; p += (size_t)N * HD * sizeof(_Float16);
  _Float16* h1h  = (_Float16*)p; p += (size_t)N * HD * sizeof(_Float16);
  uint*     hidp = (uint*)p;     p += (size_t)E * HID_STRIDE * sizeof(uint);
  _Float16* wpack= (_Float16*)p; p += (size_t)L * WPACK_PER_LAYER * sizeof(_Float16);
  float*    deg  = (float*)p;

  hipMemsetAsync(deg, 0, (size_t)N * sizeof(float), stream);
  k_encoder<<<(N + 255) / 256, 256, 0, stream>>>(x, enc_w, enc_b, h0, h0h, N);
  k_deg<<<(E + 255) / 256, 256, 0, stream>>>(edge_index, deg, E);
  const int packTotal = L * WPACK_PER_LAYER;
  k_pack<<<(packTotal + 255) / 256, 256, 0, stream>>>(w2, b2, wpack, packTotal);

  const int ngrp   = (E + (16 * TILES - 1)) / (16 * TILES);
  int nblk = (ngrp + 7) / 8;            // ~1 group per wave
  if (nblk > 1024) nblk = 1024;

  const float*    hin  = h0;  _Float16* hinh  = h0h;
  float*          hout = h1;  _Float16* houth = h1h;
  for (int l = 0; l < L; ++l) {
    hipMemsetAsync(agg, 0, (size_t)N * HD * sizeof(float), stream);
    k_edgemlp<<<(E + 255) / 256, 256, 0, stream>>>(edge_attr,
        w1 + (size_t)l * 4 * HD, b1 + (size_t)l * HD, hidp, E);
    k_edge_gemm<<<nblk, 256, 0, stream>>>(hinh, edge_index, hidp,
        wpack + (size_t)l * WPACK_PER_LAYER, agg, E);
    k_update<<<(N + 255) / 256, 256, 0, stream>>>(hin, agg, deg,
        root_w + (size_t)l * HD * HD, conv_b + (size_t)l * HD, hout, houth, N);
    float* tf = (float*)hin; hin = hout; hout = tf;
    _Float16* th = hinh; hinh = houth; houth = th;
  }
  k_out<<<(N + 255) / 256, 256, 0, stream>>>(hin, out_w, out_b,
                                             (float*)d_out, N, OD);
}
